// v2s_net_48868137894271
// MI455X (gfx1250) — matrix operand order repaired and checked by reading, not compiled, not run on hardware
//
#include <hip/hip_runtime.h>

// ---------------------------------------------------------------------------
// Fused banmo/NeRF coarse render for MI455X (gfx1250, wave32, WMMA + async).
// Kernel 1 packs all MLP weights to f16 in d_ws, pre-swizzled into the exact
// N-major LDS tile image consumed by the WMMA B-fragment loader.
// Kernel 2: one workgroup = one ray (128 samples). Activations live in LDS;
// weight chunks are DMA'd L2->LDS with global_load_async_to_lds_b64
// (ASYNCcnt-tracked, double-buffered, pipelined across all 39 chunks of the
// network) and consumed by v_wmma_f32_16x16x32_f16 with f32 accumulation.
// ---------------------------------------------------------------------------

typedef __attribute__((ext_vector_type(16))) _Float16 v16h;
typedef __attribute__((ext_vector_type(8)))  float    v8f;

namespace {

constexpr int   kThreads  = 512;                  // 16 waves
constexpr int   kChunkK   = 64;                   // K rows per weight chunk
constexpr int   kStrideH  = 264;                  // halves; 528B (16B-aligned)
constexpr int   kStrideW  = 72;                   // halves; 144B (16B-aligned)
constexpr int   kSlotB    = 256 * kStrideW * 2;   // 36864 B per packed chunk
constexpr int   kNumChunks = 39;
constexpr float kNear = 0.2f, kFar = 3.0f;

struct Params {
  const float *rays_o, *rays_d, *xys, *Rtgt, *Ttgt, *Ktgt;
  const float *w0, *b0, *w1, *b1, *w2, *b2, *w3, *b3;
  const float *w4, *b4, *w5, *b5, *w6, *b6, *w7, *b7;
  const float *w_sig, *b_sig, *w_feat, *b_feat;
  const float *w_dir, *b_dir, *w_rgb, *b_rgb;
  char* ws;      // packed f16 weight chunks (kNumChunks * kSlotB bytes)
  float* out;
};

// ---------------------------------------------------------------------------
// Chunk schedule (shared between the packer and the consumer):
//  g=0        : w0     rows   0..62  (A = posenc ex, K=64)
//  g=1..4     : w1     rows  64c..   (dense 256)
//  g=5..8     : w2          9..12: w3
//  g=13       : w4     rows   0..62  (skip: A = ex)
//  g=14..17   : w4     rows  63..318 (skip: A = h)
//  g=18..21: w5   22..25: w6   26..29: w7   30..33: w_feat
//  g=34..37   : w_dir  rows   0..255 (A = feat, 128 cols)
//  g=38       : w_dir  rows 256..282 (A = posenc dir, 128 cols)
// ---------------------------------------------------------------------------
__global__ void __launch_bounds__(256, 4) pack_weights(Params p) {
  const int g = blockIdx.x;
  const float* W; int krow0, ktot, ncols = 256;
  if (g == 0)       { W = p.w0;     krow0 = 0;               ktot = 63;  }
  else if (g < 5)   { W = p.w1;     krow0 = 64 * (g - 1);    ktot = 256; }
  else if (g < 9)   { W = p.w2;     krow0 = 64 * (g - 5);    ktot = 256; }
  else if (g < 13)  { W = p.w3;     krow0 = 64 * (g - 9);    ktot = 256; }
  else if (g == 13) { W = p.w4;     krow0 = 0;               ktot = 63;  }
  else if (g < 18)  { W = p.w4;     krow0 = 63 + 64*(g - 14); ktot = 319; }
  else if (g < 22)  { W = p.w5;     krow0 = 64 * (g - 18);   ktot = 256; }
  else if (g < 26)  { W = p.w6;     krow0 = 64 * (g - 22);   ktot = 256; }
  else if (g < 30)  { W = p.w7;     krow0 = 64 * (g - 26);   ktot = 256; }
  else if (g < 34)  { W = p.w_feat; krow0 = 64 * (g - 30);   ktot = 256; }
  else if (g < 38)  { W = p.w_dir;  krow0 = 64 * (g - 34);   ktot = 283; ncols = 128; }
  else              { W = p.w_dir;  krow0 = 256;             ktot = 283; ncols = 128; }

  _Float16* slot = (_Float16*)(p.ws + (size_t)g * kSlotB);
  // N-major image: slot[n*72 + k]; k in [0,64) valid, [64,72) zero padding.
  for (int i = threadIdx.x; i < 256 * kStrideW; i += 256) {
    const int k = i / 256;           // 0..71 (coalesced global reads over n)
    const int n = i - k * 256;       // 0..255
    float v = 0.0f;
    if (k < kChunkK && n < ncols) {
      const int gk = krow0 + k;
      if (gk < ktot) v = W[gk * ncols + n];
    }
    slot[n * kStrideW + k] = (_Float16)v;
  }
}

// ---- CDNA5 async global->LDS DMA (ASYNCcnt-tracked) -----------------------
__device__ __forceinline__ unsigned to_lds_off(const void* p) {
  // generic pointers to LDS carry the byte offset in the low 32 bits
  return (unsigned)(unsigned long long)p;
}
// Copy one 36864B packed chunk slot: 9 x b64 per lane => ASYNCcnt += 9 / wave.
__device__ __forceinline__ void copy_chunk_async(unsigned dstOff,
                                                 const char* wsBase, int g,
                                                 int tid) {
  const char* src = wsBase + (size_t)g * kSlotB + tid * 8;
  unsigned dst = dstOff + tid * 8;
#pragma unroll
  for (int j = 0; j < 9; ++j)
    asm volatile("global_load_async_to_lds_b64 %0, %1, off"
                 :: "v"(dst + j * 4096), "v"(src + j * 4096) : "memory");
}
template <int N>
__device__ __forceinline__ void wait_async() {
  asm volatile("s_wait_asynccnt %0" :: "n"(N) : "memory");
}

// ---- WMMA fragment loads (documented CDNA5 16-bit A / B VGPR layouts) -----
__device__ __forceinline__ v16h load_afrag(const _Float16* sA, int stride,
                                           int row16, int k, int lane) {
  const int r    = row16 + (lane & 15);
  const int koff = (lane < 16) ? 0 : 8;
  const _Float16* p = sA + r * stride + k + koff;
  union { v16h v; uint4 u[2]; } t;
  t.u[0] = *(const uint4*)(p);
  t.u[1] = *(const uint4*)(p + 16);
  return t.v;
}
__device__ __forceinline__ v16h load_bfrag(const _Float16* sWt, int col16,
                                           int kk, int lane) {
  const int c    = col16 + (lane & 15);
  const int koff = (lane < 16) ? 0 : 8;
  const _Float16* p = sWt + c * kStrideW + kk + koff;
  union { v16h v; uint4 u[2]; } t;
  t.u[0] = *(const uint4*)(p);
  t.u[1] = *(const uint4*)(p + 16);
  return t.v;
}

template <int MSUB, int NSUB>
__device__ __forceinline__ void mma_chunk(const _Float16* sA, int strideA,
                                          int rowBase, int k0,
                                          const _Float16* sWt, int colBase,
                                          v8f (&acc)[MSUB][NSUB], int lane) {
#pragma unroll
  for (int kk = 0; kk < kChunkK; kk += 32) {
    v16h a[MSUB];
#pragma unroll
    for (int m = 0; m < MSUB; ++m)
      a[m] = load_afrag(sA, strideA, rowBase + 16 * m, k0 + kk, lane);
#pragma unroll
    for (int n = 0; n < NSUB; ++n) {
      v16h b = load_bfrag(sWt, colBase + 16 * n, kk, lane);
#pragma unroll
      for (int m = 0; m < MSUB; ++m)
        acc[m][n] = __builtin_amdgcn_wmma_f32_16x16x32_f16(
            false, a[m], false, b, (short)0, acc[m][n], false, false);
    }
  }
}

__device__ __forceinline__ void load_bias(float* sB, const float* B, int n,
                                          int tid) {
  for (int i = tid; i < n; i += kThreads) sB[i] = B[i];
}

// C/D f32 layout: VGPR r -> M = r + (lane<16 ? 0 : 8), N = lane&15.
template <int MSUB, int NSUB>
__device__ __forceinline__ void store_acc(_Float16* sOut, int strideO,
                                          int rowBase, int colBase,
                                          v8f (&acc)[MSUB][NSUB],
                                          const float* sB, bool relu,
                                          int lane) {
  const int mofs = (lane < 16) ? 0 : 8;
  const int nofs = lane & 15;
#pragma unroll
  for (int m = 0; m < MSUB; ++m)
#pragma unroll
    for (int n = 0; n < NSUB; ++n)
#pragma unroll
      for (int r = 0; r < 8; ++r) {
        const int row = rowBase + 16 * m + r + mofs;
        const int col = colBase + 16 * n + nofs;
        float v = acc[m][n][r] + sB[col];
        if (relu) v = fmaxf(v, 0.0f);
        sOut[row * strideO + col] = (_Float16)v;
      }
}

__global__ void __launch_bounds__(kThreads, 1) nerf_fused(Params p) {
  __shared__ __align__(16) _Float16 sX[128 * 64];          // posenc(xyz)
  __shared__ __align__(16) _Float16 sH[128 * kStrideH];    // activations
  __shared__ __align__(16) _Float16 sW[2][256 * kStrideW]; // wt chunks (dbuf)
  __shared__ __align__(16) _Float16 sEd[64];               // posenc(dir), 1 row
  __shared__ float sBias[256];
  __shared__ float sSig[128];
  __shared__ float sWgt[128];
  __shared__ float sSil;
  __shared__ float sRed[128 * 6];

  const int tid  = threadIdx.x;
  const int lane = tid & 31;
  const int wid  = tid >> 5;
  const int rowBase  = (wid >> 2) * 32;   // 256-wide layers: 4x4 wave grid
  const int colBase  = (wid & 3) * 64;
  const int rowBaseD = (wid >> 1) * 16;   // 128-wide dir layer: 8x2 wave grid
  const int colBaseD = (wid & 1) * 64;

  const unsigned swOff[2] = {to_lds_off(&sW[0][0]), to_lds_off(&sW[1][0])};
  int par = 0;

  // Kick off the first weight DMA before doing any math: posenc hides it.
  copy_chunk_async(swOff[0], p.ws, 0, tid);

  const int ray = blockIdx.x;             // 0..2047, one ray per workgroup
  const int b   = ray >> 10;

  // --- posenc(xyz) per sample + posenc(dir) once per ray ------------------
  float px = 0.f, py = 0.f, pz = 0.f;
  if (tid < 128) {
    const int s = tid;
    const float ox = p.rays_o[ray*3+0], oy = p.rays_o[ray*3+1], oz = p.rays_o[ray*3+2];
    const float dx = p.rays_d[ray*3+0], dy = p.rays_d[ray*3+1], dz = p.rays_d[ray*3+2];
    const float zv = kNear + (kFar - kNear) * (float)s * (1.0f / 127.0f);
    px = ox + dx*zv; py = oy + dy*zv; pz = oz + dz*zv;
    _Float16* xr = sX + s * 64;
    xr[0] = (_Float16)px; xr[1] = (_Float16)py; xr[2] = (_Float16)pz;
    float f = 1.0f;
    for (int q = 0; q < 10; ++q) {
      const int base = 3 + 6 * q;
      xr[base+0] = (_Float16)__sinf(f*px); xr[base+1] = (_Float16)__sinf(f*py);
      xr[base+2] = (_Float16)__sinf(f*pz); xr[base+3] = (_Float16)__cosf(f*px);
      xr[base+4] = (_Float16)__cosf(f*py); xr[base+5] = (_Float16)__cosf(f*pz);
      f *= 2.0f;
    }
    xr[63] = (_Float16)0.0f;
    if (s == 0) {  // dir encoding identical for all samples: one LDS row
      const float inv = rsqrtf(dx*dx + dy*dy + dz*dz);
      const float vx = dx*inv, vy = dy*inv, vz = dz*inv;
      sEd[0] = (_Float16)vx; sEd[1] = (_Float16)vy; sEd[2] = (_Float16)vz;
      float g = 1.0f;
      for (int q = 0; q < 4; ++q) {
        const int base = 3 + 6 * q;
        sEd[base+0] = (_Float16)__sinf(g*vx); sEd[base+1] = (_Float16)__sinf(g*vy);
        sEd[base+2] = (_Float16)__sinf(g*vz); sEd[base+3] = (_Float16)__cosf(g*vx);
        sEd[base+4] = (_Float16)__cosf(g*vy); sEd[base+5] = (_Float16)__cosf(g*vz);
        g *= 2.0f;
      }
      for (int c = 27; c < 64; ++c) sEd[c] = (_Float16)0.0f;
    }
  }

  // One pipeline step: prefetch chunk g+1 into the idle buffer, wait for
  // chunk g (9 newer async ops may stay outstanding), barrier, WMMA, barrier.
  auto step = [&](int g, const _Float16* sA, int strideA, int rowB, int colB,
                  int k0, auto& acc) {
    if (g + 1 < kNumChunks) {
      copy_chunk_async(swOff[par ^ 1], p.ws, g + 1, tid);
      wait_async<9>();
    } else {
      wait_async<0>();
    }
    __syncthreads();
    mma_chunk(sA, strideA, rowB, k0, sW[par], colB, acc, lane);
    __syncthreads();
    par ^= 1;
  };

  v8f z = {0.f, 0.f, 0.f, 0.f, 0.f, 0.f, 0.f, 0.f};

  // --- layer 0: ex @ w0, relu (chunk 0) -----------------------------------
  {
    v8f acc[2][4];
#pragma unroll
    for (int m = 0; m < 2; ++m)
#pragma unroll
      for (int n = 0; n < 4; ++n) acc[m][n] = z;
    load_bias(sBias, p.b0, 256, tid);
    step(0, sX, 64, rowBase, colBase, 0, acc);
    store_acc(sH, kStrideH, rowBase, colBase, acc, sBias, true, lane);
    __syncthreads();
  }

  auto dense = [&](int base, const float* bias, bool relu) {
    v8f acc[2][4];
#pragma unroll
    for (int m = 0; m < 2; ++m)
#pragma unroll
      for (int n = 0; n < 4; ++n) acc[m][n] = z;
    load_bias(sBias, bias, 256, tid);
    for (int c = 0; c < 4; ++c)
      step(base + c, sH, kStrideH, rowBase, colBase, 64 * c, acc);
    store_acc(sH, kStrideH, rowBase, colBase, acc, sBias, relu, lane);
    __syncthreads();
  };

  dense(1, p.b1, true);   // layers 1..3
  dense(5, p.b2, true);
  dense(9, p.b3, true);

  // --- layer 4 (skip): ex part (chunk 13) + h part (chunks 14..17) --------
  {
    v8f acc[2][4];
#pragma unroll
    for (int m = 0; m < 2; ++m)
#pragma unroll
      for (int n = 0; n < 4; ++n) acc[m][n] = z;
    load_bias(sBias, p.b4, 256, tid);
    step(13, sX, 64, rowBase, colBase, 0, acc);
    for (int c = 0; c < 4; ++c)
      step(14 + c, sH, kStrideH, rowBase, colBase, 64 * c, acc);
    store_acc(sH, kStrideH, rowBase, colBase, acc, sBias, true, lane);
    __syncthreads();
  }

  dense(18, p.b5, true);  // layers 5..7
  dense(22, p.b6, true);
  dense(26, p.b7, true);

  // --- sigma head (256 -> 1), per-sample VALU dot (feat DMA in flight) ----
  load_bias(sBias, p.w_sig, 256, tid);
  __syncthreads();
  if (tid < 128) {
    float acc = p.b_sig[0];
    const _Float16* hr = sH + tid * kStrideH;
    for (int k = 0; k < 256; ++k) acc += (float)hr[k] * sBias[k];
    sSig[tid] = acc;
  }
  __syncthreads();

  dense(30, p.b_feat, false);  // feat head, in place

  // --- dir branch: feat (chunks 34..37) + ed (chunk 38), relu -> 128 ------
  {
    v8f acc[1][4];
#pragma unroll
    for (int n = 0; n < 4; ++n) acc[0][n] = z;
    load_bias(sBias, p.b_dir, 128, tid);
    for (int c = 0; c < 4; ++c)
      step(34 + c, sH, kStrideH, rowBaseD, colBaseD, 64 * c, acc);
    step(38, sEd, /*strideA=*/0, rowBaseD, colBaseD, 0, acc);  // broadcast row
    store_acc(sH, kStrideH, rowBaseD, colBaseD, acc, sBias, true, lane);
    __syncthreads();
  }

  // --- rgb head (128 -> 3) + sigmoid, kept in registers -------------------
  for (int i = tid; i < 384; i += kThreads) sRed[i] = p.w_rgb[i];
  __syncthreads();
  float rgb0 = 0.f, rgb1 = 0.f, rgb2 = 0.f;
  if (tid < 128) {
    const _Float16* hd = sH + tid * kStrideH;
    float r0 = p.b_rgb[0], r1 = p.b_rgb[1], r2 = p.b_rgb[2];
    for (int k = 0; k < 128; ++k) {
      const float hv = (float)hd[k];
      r0 += hv * sRed[k*3+0]; r1 += hv * sRed[k*3+1]; r2 += hv * sRed[k*3+2];
    }
    rgb0 = 1.0f/(1.0f+__expf(-r0));
    rgb1 = 1.0f/(1.0f+__expf(-r1));
    rgb2 = 1.0f/(1.0f+__expf(-r2));
  }
  __syncthreads();

  // --- volumetric weights: serial cumprod along the ray -------------------
  if (tid == 0) {
    const float dzv = (kFar - kNear) * (1.0f / 127.0f);
    float T = 1.0f, sil = 0.0f;
    for (int s = 0; s < 128; ++s) {
      const float sg    = fmaxf(sSig[s], 0.0f);
      const float delta = (s == 127) ? 1e10f : dzv;
      const float alpha = 1.0f - __expf(-delta * sg);
      const float w     = alpha * T;
      sWgt[s] = w;
      sil += w;
      T *= (1.0f - alpha + 1e-10f);
    }
    sSil = sil;
  }
  __syncthreads();

  // --- per-sample composited contributions (img, sil, flow) ---------------
  if (tid < 128) {
    const int s = tid;
    const float w  = sWgt[s];
    const float wn = w / (1e-9f + sSil);
    const float* R  = p.Rtgt + b * 9;
    const float* T_ = p.Ttgt + b * 3;
    const float* K_ = p.Ktgt + b * 4;
    const float xc0 = R[0]*px + R[1]*py + R[2]*pz + T_[0];
    const float xc1 = R[3]*px + R[4]*py + R[5]*pz + T_[1];
    const float xc2 = R[6]*px + R[7]*py + R[8]*pz + T_[2];
    const float Z = fmaxf(xc2, 1e-6f);
    const float u = K_[0]*xc0/Z + K_[2];
    const float v = K_[1]*xc1/Z + K_[3];
    const float xsx = p.xys[ray*2+0], xsy = p.xys[ray*2+1];
    sRed[s*6+0] = w * rgb0;
    sRed[s*6+1] = w * rgb1;
    sRed[s*6+2] = w * rgb2;
    sRed[s*6+3] = w;
    sRed[s*6+4] = wn * (u - xsx);
    sRed[s*6+5] = wn * (v - xsy);
  }
  __syncthreads();
  if (tid < 6) {
    float acc = 0.0f;
    for (int s = 0; s < 128; ++s) acc += sRed[s*6+tid];
    if (tid >= 4) acc *= (2.0f / 512.0f);   // flow scaled by /IMG_SIZE*2
    p.out[ray*6+tid] = acc;
  }
}

}  // namespace

extern "C" void kernel_launch(void* const* d_in, const int* in_sizes, int n_in,
                              void* d_out, int out_size, void* d_ws,
                              size_t ws_size, hipStream_t stream) {
  (void)in_sizes; (void)n_in; (void)out_size; (void)ws_size;
  Params p;
  p.rays_o = (const float*)d_in[0];  p.rays_d = (const float*)d_in[1];
  p.xys    = (const float*)d_in[2];  p.Rtgt   = (const float*)d_in[3];
  p.Ttgt   = (const float*)d_in[4];  p.Ktgt   = (const float*)d_in[5];
  p.w0 = (const float*)d_in[6];   p.b0 = (const float*)d_in[7];
  p.w1 = (const float*)d_in[8];   p.b1 = (const float*)d_in[9];
  p.w2 = (const float*)d_in[10];  p.b2 = (const float*)d_in[11];
  p.w3 = (const float*)d_in[12];  p.b3 = (const float*)d_in[13];
  p.w4 = (const float*)d_in[14];  p.b4 = (const float*)d_in[15];
  p.w5 = (const float*)d_in[16];  p.b5 = (const float*)d_in[17];
  p.w6 = (const float*)d_in[18];  p.b6 = (const float*)d_in[19];
  p.w7 = (const float*)d_in[20];  p.b7 = (const float*)d_in[21];
  p.w_sig  = (const float*)d_in[22]; p.b_sig  = (const float*)d_in[23];
  p.w_feat = (const float*)d_in[24]; p.b_feat = (const float*)d_in[25];
  p.w_dir  = (const float*)d_in[26]; p.b_dir  = (const float*)d_in[27];
  p.w_rgb  = (const float*)d_in[28]; p.b_rgb  = (const float*)d_in[29];
  p.ws  = (char*)d_ws;
  p.out = (float*)d_out;
  pack_weights<<<kNumChunks, 256, 0, stream>>>(p);   // one-shot f16 repack
  nerf_fused<<<2 * 1024, kThreads, 0, stream>>>(p);  // one WG per ray
}
